// GlobalCrossAttention_20444044329787
// MI455X (gfx1250) — matrix operand order repaired and checked by reading, not compiled, not run on hardware
//
#include <hip/hip_runtime.h>

typedef __attribute__((ext_vector_type(16))) _Float16 v16h;
typedef __attribute__((ext_vector_type(8)))  float    v8f;

#define NQ   300
#define NQP  304      // padded to 19 full 16-row tiles
#define CD   256
#define NH   8
#define HD   32
#define HWK  1024
#define HIDD 512

// ---- WMMA fragment index helpers (wave32, CDNA5 ISA layouts) ----
// A (16x32 f16): lane 0-15 -> M=lane, K in {0..7,16..23}; lane 16-31 -> M=lane-16, K in {8..15,24..31}
__device__ __forceinline__ int a_kidx(int lane, int hi) {
    return ((lane >> 4) << 3) + hi + ((hi >= 8) ? 8 : 0);
}
// B (32x16 f16): N = lane&15, K same as A: lane 0-15 -> {0..7,16..23}; lane 16-31 -> {8..15,24..31}
__device__ __forceinline__ int b_kidx(int lane, int hi) {
    return ((lane >> 4) << 3) + hi + ((hi >= 8) ? 8 : 0);
}
// C/D (16x16 f32): N = lane&15, VGPR r -> M = r + (lane>=16 ? 8 : 0)
__device__ __forceinline__ int c_row(int lane, int r) {
    return r + ((lane >> 4) << 3);
}

__device__ __forceinline__ v8f wmma16(v16h a, v16h b, v8f c) {
    return __builtin_amdgcn_wmma_f32_16x16x32_f16(false, a, false, b, (short)0, c, false, false);
}

// ============================================================================
// 1) CPB bias MLP (dominant compute, ~2.5 GFLOP on the matrix pipe):
//    bias[h,q,k] = sum_f relu(dx*W1[f,0]+dy*W1[f,1]+b1[f]) * W2[h,f]
//    Each wave processes 4 (q,k)-tiles: W1/b1/W2 LDS reads and the W2
//    B-fragment are loaded ONCE per K-step and reused by 4 chained WMMAs.
// ============================================================================
__global__ void __launch_bounds__(256)
cpb_bias_kernel(const float* __restrict__ ref,   // [NQ,2]
                const float* __restrict__ W1,    // [HIDD,2]
                const float* __restrict__ b1,    // [HIDD]
                const float* __restrict__ W2,    // [NH,HIDD]
                float* __restrict__ bias)        // [NH,NQ,HWK]
{
    __shared__ float    w1s[HIDD * 2];
    __shared__ float    b1s[HIDD];
    __shared__ _Float16 w2s[16 * HIDD];          // rows 8..15 zero-padded

    const int tid = threadIdx.x;
    // gfx1250 prefetch path for the weight tables we are about to stage
    __builtin_prefetch(W1, 0, 3);
    __builtin_prefetch(W2, 0, 3);
    for (int i = tid; i < HIDD * 2; i += 256) w1s[i] = W1[i];
    for (int i = tid; i < HIDD; i += 256)     b1s[i] = b1[i];
    for (int i = tid; i < 16 * HIDD; i += 256)
        w2s[i] = (i < NH * HIDD) ? (_Float16)W2[i] : (_Float16)0.0f;
    __syncthreads();

    const int lane = tid & 31;
    const int l15  = lane & 15;
    const int wave = tid >> 5;
    // 19200 16-row tiles total; 4 consecutive k-tiles per wave -> 4800 wave-jobs
    const int wt = blockIdx.x * 8 + wave;
    const int q     = wt >> 4;          // 16 wave-jobs per query
    const int kbase = (wt & 15) << 6;   // 64 keys per wave-job

    float dx[4], dy[4];
#pragma unroll
    for (int j = 0; j < 4; ++j) {
        const int k  = kbase + j * 16 + l15;    // M-row of subtile j
        const int kx = k & 31, ky = k >> 5;
        dx[j] = ref[q * 2 + 0] - (kx + 0.5f) * (1.0f / 32.0f);
        dy[j] = ref[q * 2 + 1] - (ky + 0.5f) * (1.0f / 32.0f);
    }

    v8f c0 = {}, c1 = {}, c2 = {}, c3 = {};
    const _Float16* w2row = &w2s[l15 * HIDD];

    for (int kt = 0; kt < HIDD / 32; ++kt) {
        v16h a0, a1, a2, a3, b;
#pragma unroll
        for (int hi = 0; hi < 16; ++hi) {
            const int fa = kt * 32 + a_kidx(lane, hi);
            const float2 w12 = ((const float2*)w1s)[fa];   // shared by all 4 tiles
            const float  bb  = b1s[fa];
            a0[hi] = (_Float16)fmaxf(dx[0] * w12.x + dy[0] * w12.y + bb, 0.0f);
            a1[hi] = (_Float16)fmaxf(dx[1] * w12.x + dy[1] * w12.y + bb, 0.0f);
            a2[hi] = (_Float16)fmaxf(dx[2] * w12.x + dy[2] * w12.y + bb, 0.0f);
            a3[hi] = (_Float16)fmaxf(dx[3] * w12.x + dy[3] * w12.y + bb, 0.0f);
            b[hi]  = w2row[kt * 32 + b_kidx(lane, hi)];    // shared by all 4 tiles
        }
        c0 = wmma16(a0, b, c0);
        c1 = wmma16(a1, b, c1);
        c2 = wmma16(a2, b, c2);
        c3 = wmma16(a3, b, c3);
    }

    const int h = l15;                  // N dimension = head
    if (h < NH) {
        float* bq = &bias[((size_t)h * NQ + q) * HWK];
#pragma unroll
        for (int r = 0; r < 8; ++r) {
            const int kr = c_row(lane, r);
            bq[kbase +      kr] = c0[r];
            bq[kbase + 16 + kr] = c1[r];
            bq[kbase + 32 + kr] = c2[r];
            bq[kbase + 48 + kr] = c3[r];
        }
    }
}

// ============================================================================
// 2a) Q projection: q = (raw_query + query_pos) @ Wq^T + bq  -> f16 [h][row][d]
// ============================================================================
__global__ void __launch_bounds__(128)
proj_q_kernel(const float* __restrict__ rq, const float* __restrict__ qp,
              const float* __restrict__ Wq, const float* __restrict__ bq,
              _Float16* __restrict__ qf16)   // [NH][NQP][HD]
{
    const int lane = threadIdx.x & 31;
    const int tile = blockIdx.x * 4 + (threadIdx.x >> 5);  // 19*16 = 304
    const int mt = tile >> 4, nt = tile & 15;
    const int rowq = mt * 16 + (lane & 15);
    const bool valid = rowq < NQ;

    v8f c = {};
    for (int kt = 0; kt < CD / 32; ++kt) {
        v16h a, b;
#pragma unroll
        for (int hi = 0; hi < 16; ++hi) {
            const int ka = kt * 32 + a_kidx(lane, hi);
            a[hi] = valid ? (_Float16)(rq[rowq * CD + ka] + qp[rowq * CD + ka])
                          : (_Float16)0.0f;
            const int kb = kt * 32 + b_kidx(lane, hi);
            b[hi] = (_Float16)Wq[(nt * 16 + (lane & 15)) * CD + kb];
        }
        c = wmma16(a, b, c);
    }
    const int col = nt * 16 + (lane & 15);
    const int h = col >> 5, d = col & 31;
#pragma unroll
    for (int r = 0; r < 8; ++r) {
        const int row = mt * 16 + c_row(lane, r);
        qf16[((size_t)h * NQP + row) * HD + d] = (_Float16)(c[r] + bq[col]);
    }
}

// ============================================================================
// 2b) K,V projections from memory = raw_src + src_pos  -> f16 [h][key][d]
// ============================================================================
__global__ void __launch_bounds__(128)
proj_kv_kernel(const float* __restrict__ rs, const float* __restrict__ sp,
               const float* __restrict__ Wk, const float* __restrict__ bk,
               const float* __restrict__ Wv, const float* __restrict__ bv,
               _Float16* __restrict__ kf16, _Float16* __restrict__ vf16)
{
    const int lane = threadIdx.x & 31;
    const int tile = blockIdx.x * 4 + (threadIdx.x >> 5);  // 64*16 = 1024
    const int mt = tile >> 4, nt = tile & 15;
    const int row = mt * 16 + (lane & 15);

    v8f ck = {}, cv = {};
    for (int kt = 0; kt < CD / 32; ++kt) {
        v16h a, bK, bV;
#pragma unroll
        for (int hi = 0; hi < 16; ++hi) {
            const int ka = kt * 32 + a_kidx(lane, hi);
            a[hi] = (_Float16)(rs[row * CD + ka] + sp[row * CD + ka]);
            const int kb = kt * 32 + b_kidx(lane, hi);
            const int n = nt * 16 + (lane & 15);
            bK[hi] = (_Float16)Wk[n * CD + kb];
            bV[hi] = (_Float16)Wv[n * CD + kb];
        }
        ck = wmma16(a, bK, ck);
        cv = wmma16(a, bV, cv);
    }
    const int col = nt * 16 + (lane & 15);
    const int h = col >> 5, d = col & 31;
#pragma unroll
    for (int r = 0; r < 8; ++r) {
        const int kr = mt * 16 + c_row(lane, r);
        kf16[((size_t)h * HWK + kr) * HD + d] = (_Float16)(ck[r] + bk[col]);
        vf16[((size_t)h * HWK + kr) * HD + d] = (_Float16)(cv[r] + bv[col]);
    }
}

// ============================================================================
// 3) Fused attention, key-split flash softmax:
//    4 waves per (head, 16-query tile); each wave owns 256 keys with private
//    online-softmax state, then a log-sum-exp merge through LDS.
// ============================================================================
__global__ void __launch_bounds__(128)
attention_kernel(const _Float16* __restrict__ qf16,
                 const _Float16* __restrict__ kf16,
                 const _Float16* __restrict__ vf16,
                 const float*    __restrict__ bias,
                 _Float16* __restrict__ attnout)  // [NQP][CD]
{
    __shared__ _Float16 psm[4][16 * 34];   // per-wave P transpose buffer
    __shared__ float    comb[4][32][32];   // per-wave (m,l,o0,o1) for the merge

    const int lane = threadIdx.x & 31;
    const int wave = threadIdx.x >> 5;
    const int h  = blockIdx.x / 19;
    const int q0 = (blockIdx.x % 19) * 16;
    const float scale = 0.17677669529663687f;   // 1/sqrt(32)

    // Q A-fragment (persistent across the key loop)
    v16h aq;
#pragma unroll
    for (int hi = 0; hi < 16; ++hi) {
        const int d = a_kidx(lane, hi);
        aq[hi] = qf16[((size_t)h * NQP + q0 + (lane & 15)) * HD + d];
    }

    float mrow[8], lrow[8];
#pragma unroll
    for (int r = 0; r < 8; ++r) { mrow[r] = -1e30f; lrow[r] = 0.0f; }
    v8f o0 = {}, o1 = {};

    // this wave's 256 keys: 8 chunks of 32
    for (int kc = 0; kc < 8; ++kc) {
        const int kb  = wave * 8 + kc;
        const int k0a = kb * 32, k0b = kb * 32 + 16;

        // S = Q*K^T for two 16-key subtiles (K = HD = 32, one WMMA each)
        v16h b0, b1v;
#pragma unroll
        for (int hi = 0; hi < 16; ++hi) {
            const int d = b_kidx(lane, hi);
            b0[hi]  = kf16[((size_t)h * HWK + k0a + (lane & 15)) * HD + d];
            b1v[hi] = kf16[((size_t)h * HWK + k0b + (lane & 15)) * HD + d];
        }
        v8f z = {};
        v8f s0 = wmma16(aq, b0, z);
        v8f s1 = wmma16(aq, b1v, z);

        // scale + CPB bias, online softmax update (rows live in 16-lane groups)
#pragma unroll
        for (int r = 0; r < 8; ++r) {
            const int rowq = q0 + c_row(lane, r);
            float bA = 0.0f, bB = 0.0f;
            if (rowq < NQ) {
                bA = bias[((size_t)h * NQ + rowq) * HWK + k0a + (lane & 15)];
                bB = bias[((size_t)h * NQ + rowq) * HWK + k0b + (lane & 15)];
            }
            s0[r] = s0[r] * scale + bA;
            s1[r] = s1[r] * scale + bB;

            float t = fmaxf(s0[r], s1[r]);
            t = fmaxf(t, __shfl_xor(t, 1, 32));
            t = fmaxf(t, __shfl_xor(t, 2, 32));
            t = fmaxf(t, __shfl_xor(t, 4, 32));
            t = fmaxf(t, __shfl_xor(t, 8, 32));
            const float newm = fmaxf(mrow[r], t);
            const float corr = __expf(mrow[r] - newm);
            mrow[r] = newm;
            o0[r] *= corr; o1[r] *= corr; lrow[r] *= corr;

            const float p0 = __expf(s0[r] - newm);
            const float p1 = __expf(s1[r] - newm);
            float rs = p0 + p1;
            rs += __shfl_xor(rs, 1, 32);
            rs += __shfl_xor(rs, 2, 32);
            rs += __shfl_xor(rs, 4, 32);
            rs += __shfl_xor(rs, 8, 32);
            lrow[r] += rs;

            const int prow = c_row(lane, r);
            psm[wave][prow * 34 + (lane & 15)]      = (_Float16)p0;
            psm[wave][prow * 34 + 16 + (lane & 15)] = (_Float16)p1;
        }
        __syncthreads();   // uniform across all 4 waves (8 iterations each)

        // O += P * V  (A from LDS transpose buffer; N split over HD halves)
        v16h ap, bva, bvb;
#pragma unroll
        for (int hi = 0; hi < 16; ++hi) {
            ap[hi] = psm[wave][(lane & 15) * 34 + a_kidx(lane, hi)];
            const int key = kb * 32 + b_kidx(lane, hi);
            bva[hi] = vf16[((size_t)h * HWK + key) * HD + (lane & 15)];
            bvb[hi] = vf16[((size_t)h * HWK + key) * HD + 16 + (lane & 15)];
        }
        o0 = wmma16(ap, bva, o0);
        o1 = wmma16(ap, bvb, o1);
        __syncthreads();
    }

    // publish per-wave partial state
#pragma unroll
    for (int r = 0; r < 8; ++r) {
        comb[wave][lane][r]      = mrow[r];
        comb[wave][lane][8 + r]  = lrow[r];
        comb[wave][lane][16 + r] = o0[r];
        comb[wave][lane][24 + r] = o1[r];
    }
    __syncthreads();

    // log-sum-exp merge (wave 0): O = sum_w e^{m_w-M} O_w ; L = sum_w e^{m_w-M} l_w
    if (wave == 0) {
#pragma unroll
        for (int r = 0; r < 8; ++r) {
            float M = -1e30f;
#pragma unroll
            for (int w = 0; w < 4; ++w) M = fmaxf(M, comb[w][lane][r]);
            float L = 0.0f, O0 = 0.0f, O1 = 0.0f;
#pragma unroll
            for (int w = 0; w < 4; ++w) {
                const float cw = __expf(comb[w][lane][r] - M);
                L  += cw * comb[w][lane][8 + r];
                O0 += cw * comb[w][lane][16 + r];
                O1 += cw * comb[w][lane][24 + r];
            }
            const float inv = 1.0f / L;
            const int rowq = q0 + c_row(lane, r);
            attnout[(size_t)rowq * CD + h * HD + (lane & 15)]      = (_Float16)(O0 * inv);
            attnout[(size_t)rowq * CD + h * HD + 16 + (lane & 15)] = (_Float16)(O1 * inv);
        }
    }
}

// ============================================================================
// 4) Output projection: out = attnout @ Wo^T + bo   (f32 result)
// ============================================================================
__global__ void __launch_bounds__(128)
out_proj_kernel(const _Float16* __restrict__ attnout,  // [NQP][CD]
                const float* __restrict__ Wo, const float* __restrict__ bo,
                float* __restrict__ out)               // [NQ][CD]
{
    const int lane = threadIdx.x & 31;
    const int tile = blockIdx.x * 4 + (threadIdx.x >> 5);  // 19*16 = 304
    const int mt = tile >> 4, nt = tile & 15;

    v8f c = {};
    for (int kt = 0; kt < CD / 32; ++kt) {
        v16h a, b;
#pragma unroll
        for (int hi = 0; hi < 16; ++hi) {
            const int ka = kt * 32 + a_kidx(lane, hi);
            a[hi] = attnout[(size_t)(mt * 16 + (lane & 15)) * CD + ka];
            const int kb = kt * 32 + b_kidx(lane, hi);
            b[hi] = (_Float16)Wo[(nt * 16 + (lane & 15)) * CD + kb];
        }
        c = wmma16(a, b, c);
    }
    const int col = nt * 16 + (lane & 15);
#pragma unroll
    for (int r = 0; r < 8; ++r) {
        const int rowq = mt * 16 + c_row(lane, r);
        if (rowq < NQ) out[(size_t)rowq * CD + col] = c[r] + bo[col];
    }
}

// ============================================================================
extern "C" void kernel_launch(void* const* d_in, const int* in_sizes, int n_in,
                              void* d_out, int out_size, void* d_ws, size_t ws_size,
                              hipStream_t stream) {
    (void)in_sizes; (void)n_in; (void)out_size; (void)ws_size;

    const float* raw_query = (const float*)d_in[0];
    const float* query_pos = (const float*)d_in[1];
    const float* ref_pts   = (const float*)d_in[2];
    const float* raw_src   = (const float*)d_in[3];
    const float* src_pos   = (const float*)d_in[4];
    const float* Wq = (const float*)d_in[5];  const float* bq = (const float*)d_in[6];
    const float* Wk = (const float*)d_in[7];  const float* bk = (const float*)d_in[8];
    const float* Wv = (const float*)d_in[9];  const float* bv = (const float*)d_in[10];
    const float* Wo = (const float*)d_in[11]; const float* bo = (const float*)d_in[12];
    const float* W1 = (const float*)d_in[13]; const float* b1 = (const float*)d_in[14];
    const float* W2 = (const float*)d_in[15];
    float* out = (float*)d_out;

    // workspace layout (bytes)
    char* ws = (char*)d_ws;
    _Float16* qf16    = (_Float16*)(ws + 0);                    // 8*304*32*2   = 155648
    _Float16* kf16    = (_Float16*)(ws + 155648);               // 8*1024*32*2  = 524288
    _Float16* vf16    = (_Float16*)(ws + 679936);               // 524288
    float*    biasbuf = (float*)   (ws + 1204224);              // 8*300*1024*4 = 9830400
    _Float16* attnout = (_Float16*)(ws + 11034624);             // 304*256*2    = 155648

    // independent producers (stream-ordered)
    cpb_bias_kernel<<<600, 256, 0, stream>>>(ref_pts, W1, b1, W2, biasbuf);
    proj_q_kernel <<<76, 128, 0, stream>>>(raw_query, query_pos, Wq, bq, qf16);
    proj_kv_kernel<<<256, 128, 0, stream>>>(raw_src, src_pos, Wk, bk, Wv, bv, kf16, vf16);
    // fused attention (consumes all of the above)
    attention_kernel<<<8 * 19, 128, 0, stream>>>(qf16, kf16, vf16, biasbuf, attnout);
    // final projection
    out_proj_kernel<<<76, 128, 0, stream>>>(attnout, Wo, bo, out);
}
